// GridExtractor_62517543960968
// MI455X (gfx1250) — matrix operand order repaired and checked by reading, not compiled, not run on hardware
//
#include <hip/hip_runtime.h>
#include <hip/hip_bf16.h>
#include <math.h>

// ---- WMMA vector types (CDNA5 / gfx1250, wave32) ----
typedef __attribute__((ext_vector_type(16))) __bf16 v16bf;
typedef __attribute__((ext_vector_type(8)))  __bf16 bf16x8;
typedef __attribute__((ext_vector_type(8)))  float  v8f;

union AB16 { v16bf v; bf16x8 h[2]; };

// ---- explicit global-address-space accessors (force global_load/store, not flat) ----
#define GAS __attribute__((address_space(1)))
__device__ __forceinline__ bf16x8 gload8(const __bf16* p) {
    return *(const GAS bf16x8*)(unsigned long long)p;
}
__device__ __forceinline__ float gloadf(const float* p) {
    return *(const GAS float*)(unsigned long long)p;
}
__device__ __forceinline__ void gstoref(float* p, float v) {
    *(GAS float*)(unsigned long long)p = v;
}

#define CDIM 512
#define NSEQ 64

// LDS layout (bytes). xT padded to stride 520 (16B-aligned rows), k/q/v stride 72.
#define LDX   520
#define LDK   72
#define HSTR  (64*LDK)          // 4608 elements per head
#define OFF_X 0                 // xT[64][520] bf16 (66560 B) -> reused as wT[h][64][64]
#define OFF_K 66560
#define OFF_Q 140288
#define OFF_V 214016
#define OFF_M 287744            // mask: 64 floats
#define LDS_BYTES 288000

#define X_ELEMS  16777216       // 8*512*64*64
#define M_ELEMS  32768          // 8*64*64
#define WELEMS   (3*512*512)    // one weight tensor (L,C,C)

__global__ void __launch_bounds__(256)
axial_attn_stage(const float* __restrict__ xin,
                 float* __restrict__ xout,
                 const __bf16* __restrict__ Wk,
                 const __bf16* __restrict__ Wq,
                 const __bf16* __restrict__ Wv,
                 const float* __restrict__ bk,
                 const float* __restrict__ bq,
                 const float* __restrict__ bv,
                 const float* __restrict__ masks,
                 int isCol)
{
    extern __shared__ char smem[];
    __bf16* sX = (__bf16*)(smem + OFF_X);
    __bf16* sK = (__bf16*)(smem + OFF_K);
    __bf16* sQ = (__bf16*)(smem + OFF_Q);
    __bf16* sV = (__bf16*)(smem + OFF_V);
    float*  sM = (float*)(smem + OFF_M);

    const int wg = blockIdx.x;          // 0..4095 -> (b, r|j)
    const int b  = wg >> 6;
    const int rj = wg & 63;
    const int base      = b * (CDIM * 64 * 64) + (isCol ? rj * 64 : rj);
    const int seqStride = isCol ? 1 : 64;
    const int mbase     = b * 4096 + (isCol ? rj * 64 : rj);
    const int mstride   = isCol ? 1 : 64;

    const int tid = threadIdx.x;
    if (tid < 64) sM[tid] = gloadf(masks + mbase + tid * mstride);

    // load x slice -> LDS transposed bf16: xT[s][ch]
    for (int e = tid; e < CDIM * NSEQ; e += 256) {
        int ch = e >> 6, s = e & 63;
        float v = gloadf(xin + base + ch * 4096 + s * seqStride);
        sX[s * LDX + ch] = (__bf16)v;
    }
    __syncthreads();

    const int wave = tid >> 5;
    const int lane = tid & 31;
    const int hl   = lane >> 4;     // lane half (0/1)
    const int lr   = lane & 15;

    // ---------------- projections: k,q,v = W @ x + b ----------------
    // Each wave owns a (proj, 16-row o-strip) and computes all 4 n-tiles at once,
    // so every A (weight) fragment is fetched from L2 exactly once.
    const __bf16* Ws[3] = {Wk, Wq, Wv};
    const float*  Bs[3] = {bk, bq, bv};
    for (int strip = wave; strip < 96; strip += 8) {   // 3 proj * 32 o-strips
        int p  = strip >> 5;          // 0..2
        int ot = strip & 31;          // 0..31
        int o0 = ot * 16;
        const __bf16* W = Ws[p];
        const float* bias = Bs[p];

        v8f acc[4];
        #pragma unroll
        for (int nt = 0; nt < 4; ++nt)
            #pragma unroll
            for (int r = 0; r < 8; ++r)
                acc[nt][r] = gloadf(bias + o0 + r + 8 * hl);

        const __bf16* wrow = W + (o0 + lr) * CDIM + 8 * hl;
        #pragma unroll 2
        for (int kc = 0; kc < 16; ++kc) {
            int c0 = kc * 32;
            AB16 a;
            a.h[0] = gload8(wrow + c0);        // K = c0+(0..7 | 8..15)
            a.h[1] = gload8(wrow + c0 + 16);   // K = c0+16+(0..7 | 8..15)
            #pragma unroll
            for (int nt = 0; nt < 4; ++nt) {
                AB16 bb;
                bb.h[0] = *(const bf16x8*)(sX + (nt * 16 + lr) * LDX + c0 + 8 * hl);
                bb.h[1] = *(const bf16x8*)(sX + (nt * 16 + lr) * LDX + c0 + 16 + 8 * hl);
                acc[nt] = __builtin_amdgcn_wmma_f32_16x16x32_bf16(
                              false, a.v, false, bb.v, (short)0, acc[nt], false, false);
            }
        }
        // scatter-store with head de-interleave: o = d*8 + h
        __bf16* dst = (p == 0) ? sK : (p == 1) ? sQ : sV;
        #pragma unroll
        for (int nt = 0; nt < 4; ++nt) {
            int n = nt * 16 + lr;
            #pragma unroll
            for (int r = 0; r < 8; ++r) {
                int o = o0 + r + 8 * hl;
                int h = o & 7, d = o >> 3;
                int off = (p < 2) ? (h * HSTR + n * LDK + d)    // kT,qT: [h][n][d]
                                  : (h * HSTR + d * LDK + n);   // v:     [h][d][n]
                dst[off] = (__bf16)acc[nt][r];
            }
        }
    }
    __syncthreads();

    // ---------------- attention: wave == head ----------------
    const int h = wave;
    const __bf16* kh = sK + h * HSTR;
    const __bf16* qh = sQ + h * HSTR;
    const __bf16* vh = sV + h * HSTR;
    __bf16* wh = sX + h * 4096;     // reuse xT region: wT[qq][kk], ld=64

    // logits[kk][qq] = sum_d K[d][kk] * Q[d][qq]  (A = kT rows, B = qT rows)
    v8f L[4][4];
    #pragma unroll
    for (int mt = 0; mt < 4; ++mt) {
        #pragma unroll
        for (int nt = 0; nt < 4; ++nt) {
            v8f acc = {0.f,0.f,0.f,0.f,0.f,0.f,0.f,0.f};
            #pragma unroll
            for (int kd = 0; kd < 2; ++kd) {
                int d0 = kd * 32;
                AB16 a, bb;
                a.h[0]  = *(const bf16x8*)(kh + (mt * 16 + lr) * LDK + d0 + 8 * hl);
                a.h[1]  = *(const bf16x8*)(kh + (mt * 16 + lr) * LDK + d0 + 16 + 8 * hl);
                bb.h[0] = *(const bf16x8*)(qh + (nt * 16 + lr) * LDK + d0 + 8 * hl);
                bb.h[1] = *(const bf16x8*)(qh + (nt * 16 + lr) * LDK + d0 + 16 + 8 * hl);
                acc = __builtin_amdgcn_wmma_f32_16x16x32_bf16(
                          false, a.v, false, bb.v, (short)0, acc, false, false);
            }
            L[mt][nt] = acc;
        }
    }

    // softmax over keys kk for each column qq (lane holds one column per nt)
    const float scale = 0.044194173824159216f;   // 1/sqrt(512)
    #pragma unroll
    for (int nt = 0; nt < 4; ++nt) {
        float mx = -3.0e38f;
        #pragma unroll
        for (int mt = 0; mt < 4; ++mt)
            #pragma unroll
            for (int r = 0; r < 8; ++r) {
                int kk = mt * 16 + r + 8 * hl;
                float l = L[mt][nt][r] * scale - (1.0f - sM[kk]) * 1e8f;
                L[mt][nt][r] = l;
                mx = fmaxf(mx, l);
            }
        mx = fmaxf(mx, __shfl_xor(mx, 16, 32));   // join row-halves
        float sum = 0.f;
        #pragma unroll
        for (int mt = 0; mt < 4; ++mt)
            #pragma unroll
            for (int r = 0; r < 8; ++r) {
                float e = __expf(L[mt][nt][r] - mx);
                L[mt][nt][r] = e;
                sum += e;
            }
        sum += __shfl_xor(sum, 16, 32);
        float inv = 1.0f / sum;
        int qq = nt * 16 + lr;
        #pragma unroll
        for (int mt = 0; mt < 4; ++mt)
            #pragma unroll
            for (int r = 0; r < 8; ++r) {
                int kk = mt * 16 + r + 8 * hl;
                wh[qq * 64 + kk] = (__bf16)(L[mt][nt][r] * inv);   // transposed store
            }
    }

    // O[d][qq] = sum_kk V[d][kk] * w[kk][qq]; epilogue adds residual
    #pragma unroll
    for (int dt = 0; dt < 4; ++dt) {
        #pragma unroll
        for (int nt = 0; nt < 4; ++nt) {
            v8f acc = {0.f,0.f,0.f,0.f,0.f,0.f,0.f,0.f};
            #pragma unroll
            for (int kt = 0; kt < 2; ++kt) {
                int kk0 = kt * 32;
                AB16 a, bb;
                a.h[0]  = *(const bf16x8*)(vh + (dt * 16 + lr) * LDK + kk0 + 8 * hl);
                a.h[1]  = *(const bf16x8*)(vh + (dt * 16 + lr) * LDK + kk0 + 16 + 8 * hl);
                bb.h[0] = *(const bf16x8*)(wh + (nt * 16 + lr) * 64 + kk0 + 8 * hl);
                bb.h[1] = *(const bf16x8*)(wh + (nt * 16 + lr) * 64 + kk0 + 16 + 8 * hl);
                acc = __builtin_amdgcn_wmma_f32_16x16x32_bf16(
                          false, a.v, false, bb.v, (short)0, acc, false, false);
            }
            #pragma unroll
            for (int r = 0; r < 8; ++r) {
                int d  = dt * 16 + r + 8 * hl;
                int qq = nt * 16 + lr;
                int ch = d * 8 + h;
                int gi = base + ch * 4096 + qq * seqStride;
                gstoref(xout + gi, acc[r] + gloadf(xin + gi));
            }
        }
    }
}

__global__ void cvt_w_bf16(const float* __restrict__ src, __bf16* __restrict__ dst, int n) {
    int i = blockIdx.x * blockDim.x + threadIdx.x;
    if (i < n) dst[i] = (__bf16)gloadf(src + i);
}

__global__ void copy_f32(const float* __restrict__ src, float* __restrict__ dst, int n) {
    int i = blockIdx.x * blockDim.x + threadIdx.x;
    if (i < n) gstoref(dst + i, gloadf(src + i));
}

extern "C" void kernel_launch(void* const* d_in, const int* in_sizes, int n_in,
                              void* d_out, int out_size, void* d_ws, size_t ws_size,
                              hipStream_t stream) {
    (void)in_sizes; (void)n_in; (void)out_size; (void)ws_size;
    const float* x     = (const float*)d_in[0];
    const float* masks = (const float*)d_in[1];
    // weights in dict order: col_Wk,col_Wq,col_Wv,row_Wk,row_Wq,row_Wv
    const float* W[6];
    for (int i = 0; i < 6; ++i) W[i] = (const float*)d_in[2 + i];
    // biases: col_bk,col_bq,col_bv,row_bk,row_bq,row_bv
    const float* B[6];
    for (int i = 0; i < 6; ++i) B[i] = (const float*)d_in[8 + i];

    float* out = (float*)d_out;
    char*  ws  = (char*)d_ws;
    float*  xping = (float*)ws;                                   // 67 MB ping buffer
    __bf16* wbf   = (__bf16*)(ws + (size_t)X_ELEMS * sizeof(float));

    // per-call: f32 -> bf16 weight copy (L2-resident afterwards)
    for (int i = 0; i < 6; ++i)
        cvt_w_bf16<<<(WELEMS + 255) / 256, 256, 0, stream>>>(
            W[i], wbf + (size_t)i * WELEMS, WELEMS);

    // second output of the tuple: masks pass-through
    copy_f32<<<(M_ELEMS + 255) / 256, 256, 0, stream>>>(masks, out + X_ELEMS, M_ELEMS);

    // 6 stages: in -> ws -> out -> ws -> out -> ws -> out
    const float* src = x;
    for (int s = 0; s < 6; ++s) {
        int lay   = s >> 1;
        int isCol = ((s & 1) == 0);
        int wsel  = isCol ? 0 : 3;
        const __bf16* Wk = wbf + (size_t)(wsel + 0) * WELEMS + (size_t)lay * 512 * 512;
        const __bf16* Wq = wbf + (size_t)(wsel + 1) * WELEMS + (size_t)lay * 512 * 512;
        const __bf16* Wv = wbf + (size_t)(wsel + 2) * WELEMS + (size_t)lay * 512 * 512;
        const float* bk = B[wsel + 0] + lay * 512;
        const float* bq = B[wsel + 1] + lay * 512;
        const float* bv = B[wsel + 2] + lay * 512;
        float* dst = (s % 2 == 0) ? xping : out;
        axial_attn_stage<<<4096, 256, LDS_BYTES, stream>>>(
            src, dst, Wk, Wq, Wv, bk, bq, bv, masks, isCol);
        src = dst;
    }
}
